// ThermoCellularPINN_72782515798427
// MI455X (gfx1250) — matrix operand order repaired and checked by reading, not compiled, not run on hardware
//
#include <hip/hip_runtime.h>
#include <hip/hip_bf16.h>

// ---------------- sizes ----------------
#define S      4096
#define DE     256
#define DC     64
#define DIN    320
#define DM     64
#define DEPTH  4
#define KSPLIT 4
#define SKEYS  (S / KSPLIT)     // 1024 keys per partial
#define SCALE  0.0625f          // 1/sqrt(256)
#define DT_    0.1f
#define THRESH 0.05f

typedef __attribute__((ext_vector_type(16))) __bf16 v16bf;
typedef __attribute__((ext_vector_type(8)))  __bf16 v8bf;
typedef __attribute__((ext_vector_type(8)))  float  v8f;

// ---------------- WMMA helpers (CDNA5 wave32, per ISA layout tables) ----------------
__device__ __forceinline__ v8f wmma_bf16(v16bf a, v16bf b, v8f c) {
  return __builtin_amdgcn_wmma_f32_16x16x32_bf16(false, a, false, b, (short)0, c,
                                                 false, false);
}

// A matrix 16x32 (MxK) from row-major src[row][col], ld = row stride in elements.
__device__ __forceinline__ v16bf loadA16x32(const __bf16* src, int ld, int r0, int k0) {
  int lane = threadIdx.x & 31, m = lane & 15, half = lane >> 4;
  const __bf16* base = src + (size_t)(r0 + m) * ld + k0 + half * 8;
  v8bf lo = *(const v8bf*)(base);
  v8bf hi = *(const v8bf*)(base + 16);
  return __builtin_shufflevector(lo, hi, 0,1,2,3,4,5,6,7,8,9,10,11,12,13,14,15);
}

// B matrix 32x16 (KxN) where memory is row-major over N (src[n][k]):
// element h -> K = half*8 + h (h<8), 16 + half*8 + (h-8) (h>=8).
__device__ __forceinline__ v16bf loadB_NK(const __bf16* src, int ld, int n0, int k0) {
  int lane = threadIdx.x & 31, n = lane & 15, half = lane >> 4;
  const __bf16* base = src + (size_t)(n0 + n) * ld + k0 + half * 8;
  v8bf lo = *(const v8bf*)(base);
  v8bf hi = *(const v8bf*)(base + 16);
  return __builtin_shufflevector(lo, hi, 0,1,2,3,4,5,6,7,8,9,10,11,12,13,14,15);
}

__device__ __forceinline__ float redmax16(float v) {
  v = fmaxf(v, __shfl_xor(v, 1, 32));
  v = fmaxf(v, __shfl_xor(v, 2, 32));
  v = fmaxf(v, __shfl_xor(v, 4, 32));
  v = fmaxf(v, __shfl_xor(v, 8, 32));
  return v;
}
__device__ __forceinline__ float redsum16(float v) {
  v += __shfl_xor(v, 1, 32);
  v += __shfl_xor(v, 2, 32);
  v += __shfl_xor(v, 4, 32);
  v += __shfl_xor(v, 8, 32);
  return v;
}

// ---------------- elementwise prep / copy ----------------
__global__ void copy_f32(float* dst, const float* src, int n) {
  int i = blockIdx.x * blockDim.x + threadIdx.x;
  if (i < n) dst[i] = src[i];
}

__global__ void prep_x(__bf16* xb, const float* e, const float* c) {
  int i = blockIdx.x * blockDim.x + threadIdx.x;
  if (i >= S * DIN) return;
  int t = i / DIN, d = i % DIN;
  float v = (d < DE) ? e[(size_t)t * DE + d] : c[(size_t)t * DC + (d - DE)];
  xb[i] = (__bf16)v;
}

__global__ void prep_bf(__bf16* dst, const float* src, int n) {
  int i = blockIdx.x * blockDim.x + threadIdx.x;
  if (i < n) dst[i] = (__bf16)src[i];
}

// transposed weights: wt[sel][dout][din] = w[layer][din][dout]
__global__ void prep_w(__bf16* wt, const float* wq, const float* wk,
                       const float* wv, int layer) {
  int i = blockIdx.x * blockDim.x + threadIdx.x;
  const int per = DIN * DE;
  if (i >= 3 * per) return;
  int sel = i / per, r = i % per;
  int dout = r / DIN, din = r % DIN;
  const float* src = (sel == 0) ? wq : (sel == 1) ? wk : wv;
  wt[i] = (__bf16)src[(size_t)layer * per + (size_t)din * DE + dout];
}

// ---------------- QKV projection: [S x DIN] x [DIN x DE] ----------------
// q,k stored row-major [S][DE]; v stored transposed [DE][S] for the P@V GEMM.
__global__ __launch_bounds__(32) void qkv_kernel(const __bf16* xb, const __bf16* wt,
                                                 const float* bq, const float* bk,
                                                 const float* bv, __bf16* qb,
                                                 __bf16* kb, __bf16* vt, int layer) {
  int q0 = blockIdx.x * 16;
  int sel = blockIdx.y;
  const __bf16* w = wt + (size_t)sel * DIN * DE;   // [DE][DIN]
  const float* bias = ((sel == 0) ? bq : (sel == 1) ? bk : bv) + (size_t)layer * DE;
  int lane = threadIdx.x, half = lane >> 4, col = lane & 15;

  v16bf xA[10];
#pragma unroll
  for (int kblk = 0; kblk < 10; ++kblk)
    xA[kblk] = loadA16x32(xb, DIN, q0, kblk * 32);

  for (int nt = 0; nt < 16; ++nt) {
    v8f acc = {};
    // software-pipelined: load B-frag for kblk+1 before consuming kblk
    v16bf bcur = loadB_NK(w, DIN, nt * 16, 0);
#pragma unroll
    for (int kblk = 0; kblk < 10; ++kblk) {
      v16bf bnext = bcur;
      if (kblk < 9) bnext = loadB_NK(w, DIN, nt * 16, (kblk + 1) * 32);
      acc = wmma_bf16(xA[kblk], bcur, acc);
      bcur = bnext;
    }
    int cc = nt * 16 + col;
    float bsv = bias[cc];
    if (sel == 2) {
#pragma unroll
      for (int i = 0; i < 8; ++i)
        vt[(size_t)cc * S + (q0 + i + 8 * half)] = (__bf16)(acc[i] + bsv);
    } else {
      __bf16* dst = (sel == 0) ? qb : kb;
#pragma unroll
      for (int i = 0; i < 8; ++i)
        dst[(size_t)(q0 + i + 8 * half) * DE + cc] = (__bf16)(acc[i] + bsv);
    }
  }
}

// ---------------- masked flash attention (key-split partials) ----------------
__global__ __launch_bounds__(32) void attn_kernel(const __bf16* qb, const __bf16* kb,
                                                  const __bf16* vt, const __bf16* mb,
                                                  const float* lw, float* Opart,
                                                  float* rmaxp, float* rsump) {
  int q0 = blockIdx.x * 16;
  int part = blockIdx.y;
  int kbeg = part * SKEYS, kend = kbeg + SKEYS;
  int lane = threadIdx.x, half = lane >> 4, col = lane & 15;
  __shared__ __align__(32) __bf16 pmat[16 * 32];

  v16bf Qf[8];
#pragma unroll
  for (int f = 0; f < 8; ++f) Qf[f] = loadA16x32(qb, DE, q0, f * 32);
  v16bf Mf[2];
#pragma unroll
  for (int f = 0; f < 2; ++f) Mf[f] = loadA16x32(mb, DM, q0, f * 32);

  float ls[8];
#pragma unroll
  for (int i = 0; i < 8; ++i) ls[i] = lw[q0 + i + 8 * half];

  v8f O[16];
#pragma unroll
  for (int nt = 0; nt < 16; ++nt) O[nt] = (v8f){};
  float rmax[8], rsum[8];
#pragma unroll
  for (int i = 0; i < 8; ++i) { rmax[i] = -3.0e38f; rsum[i] = 0.0f; }

  for (int t0 = kbeg; t0 < kend; t0 += 32) {
    // gfx1250 global_prefetch_b8: next K tile + current V^T panel
    if (t0 + 32 < kend) {
      __builtin_prefetch(kb + (size_t)(t0 + 32 + lane) * DE, 0, 3);
      __builtin_prefetch(kb + (size_t)(t0 + 32 + lane) * DE + 128, 0, 3);
    }
#pragma unroll
    for (int r = 0; r < 8; ++r)   // 256 rows of V^T, one 128B line covers 2 blocks
      __builtin_prefetch(vt + (size_t)(lane * 8 + r) * S + t0, 0, 3);

    // membrane B-frags up-front (small, overlap with score loads)
    v16bf mB0[2], mB1[2];
#pragma unroll
    for (int f = 0; f < 2; ++f) {
      mB0[f] = loadB_NK(mb, DM, t0, f * 32);
      mB1[f] = loadB_NK(mb, DM, t0 + 16, f * 32);
    }

    // scores: Q [16x256] x K^T -> two 16x16 tiles (one-frag-ahead pipeline)
    v8f c0 = {}, c1 = {};
    v16bf kb0 = loadB_NK(kb, DE, t0, 0);
    v16bf kb1 = loadB_NK(kb, DE, t0 + 16, 0);
#pragma unroll
    for (int f = 0; f < 8; ++f) {
      v16bf nb0 = kb0, nb1 = kb1;
      if (f < 7) {
        nb0 = loadB_NK(kb, DE, t0, (f + 1) * 32);
        nb1 = loadB_NK(kb, DE, t0 + 16, (f + 1) * 32);
      }
      c0 = wmma_bf16(Qf[f], kb0, c0);
      c1 = wmma_bf16(Qf[f], kb1, c1);
      kb0 = nb0; kb1 = nb1;
    }
    // membrane: M [16x64] x M^T
    v8f g0 = {}, g1 = {};
#pragma unroll
    for (int f = 0; f < 2; ++f) {
      g0 = wmma_bf16(Mf[f], mB0[f], g0);
      g1 = wmma_bf16(Mf[f], mB1[f], g1);
    }
    float lt0 = lw[t0 + col], lt1 = lw[t0 + 16 + col];

    float p0[8], p1[8], fac[8];
#pragma unroll
    for (int i = 0; i < 8; ++i) {
      float a0 = c0[i] * SCALE * g0[i] * ls[i] * lt0;
      float a1 = c1[i] * SCALE * g1[i] * ls[i] * lt1;
      float mx = redmax16(fmaxf(a0, a1));
      float nm = fmaxf(rmax[i], mx);
      fac[i] = __expf(rmax[i] - nm);
      rmax[i] = nm;
      p0[i] = __expf(a0 - nm);
      p1[i] = __expf(a1 - nm);
      rsum[i] = rsum[i] * fac[i] + redsum16(p0[i] + p1[i]);
    }
#pragma unroll
    for (int nt = 0; nt < 16; ++nt)
#pragma unroll
      for (int i = 0; i < 8; ++i) O[nt][i] *= fac[i];

    __syncthreads();
#pragma unroll
    for (int i = 0; i < 8; ++i) {
      pmat[(i + 8 * half) * 32 + col]      = (__bf16)p0[i];
      pmat[(i + 8 * half) * 32 + col + 16] = (__bf16)p1[i];
    }
    __syncthreads();
    v16bf Pa = loadA16x32(pmat, 32, 0, 0);
    // P [16x32] x V [32 keys x 16 feats], V^T stored [DE][S] -> contiguous B
    v16bf vb = loadB_NK(vt, S, 0, t0);
#pragma unroll
    for (int nt = 0; nt < 16; ++nt) {
      v16bf nv = vb;
      if (nt < 15) nv = loadB_NK(vt, S, (nt + 1) * 16, t0);
      O[nt] = wmma_bf16(Pa, vb, O[nt]);
      vb = nv;
    }
  }

  // write unnormalized partial + running stats
  float* Op = Opart + (size_t)part * S * DE;
#pragma unroll
  for (int nt = 0; nt < 16; ++nt)
#pragma unroll
    for (int i = 0; i < 8; ++i)
      Op[(size_t)(q0 + i + 8 * half) * DE + nt * 16 + col] = O[nt][i];
  if (col == 0) {
#pragma unroll
    for (int i = 0; i < 8; ++i) {
      rmaxp[part * S + q0 + i + 8 * half] = rmax[i];
      rsump[part * S + q0 + i + 8 * half] = rsum[i];
    }
  }
}

// combine key-split partials: h = sum_p O_p*exp(m_p-M) / sum_p s_p*exp(m_p-M)
__global__ void merge_attn(float* h, const float* Opart, const float* rmaxp,
                           const float* rsump) {
  int i = blockIdx.x * blockDim.x + threadIdx.x;
  if (i >= S * DE) return;
  int row = i >> 8;
  float M = -3.0e38f;
#pragma unroll
  for (int p = 0; p < KSPLIT; ++p) M = fmaxf(M, rmaxp[p * S + row]);
  float num = 0.0f, den = 0.0f;
#pragma unroll
  for (int p = 0; p < KSPLIT; ++p) {
    float w = __expf(rmaxp[p * S + row] - M);
    num += Opart[(size_t)p * S * DE + i] * w;
    den += rsump[p * S + row] * w;
  }
  h[i] = num / den;
}

// ---------------- per-token thermodynamic ODE ----------------
__global__ __launch_bounds__(32) void ode_kernel(float* e, const float* h, float* E,
                                                 float* P, float* G, float* m,
                                                 float* l, const float* alpha,
                                                 const float* beta, const float* gamma,
                                                 const float* basal, int layer) {
  int t = blockIdx.x;
  int lane = threadIdx.x;
  const float* hr = h + (size_t)t * DE;
  float* er = e + (size_t)t * DE;
  float s = 0.0f;
#pragma unroll
  for (int j = 0; j < 8; ++j) {
    float hv = hr[lane + j * 32];
    s += hv * hv;
    er[lane + j * 32] += hv;
  }
  s += __shfl_xor(s, 16, 32);
  s += __shfl_xor(s, 8, 32);
  s += __shfl_xor(s, 4, 32);
  s += __shfl_xor(s, 2, 32);
  s += __shfl_xor(s, 1, 32);

  float Ev = E[t], Pv = P[t];
  float PE = Pv * Ev;
  float Pn = fmaxf(0.0f, Pv + (alpha[layer] * s - beta[layer] * PE) * DT_);
  float En = fmaxf(0.0f, Ev + (basal[layer] - gamma[layer] * PE) * DT_);
  float mt = 1.0f / (1.0f + __expf(-(En - Pn)));
  float Gn = G[t] + (0.1f * En - 0.5f * Pn) * DT_;
  float ldc = 0.01f + 0.5f * fmaxf(0.0f, Pn - 2.0f);
  float ln = fminf(1.0f, fmaxf(0.0f, l[t] - ldc * DT_));

  float* mr = m + (size_t)t * DM;
#pragma unroll
  for (int j = 0; j < 2; ++j)
    mr[lane + j * 32] = 0.9f * mr[lane + j * 32] + 0.1f * mt;

  if (lane == 0) { E[t] = En; P[t] = Pn; G[t] = Gn; l[t] = ln; }
}

// ---------------- finalize: alive mask + outputs ----------------
#define OUT_E  0
#define OUT_EE 1048576
#define OUT_P  1052672
#define OUT_G  1056768
#define OUT_M  1060864
#define OUT_C  1323008
#define OUT_L  1585152
#define OUT_A  1589248
#define OUT_ND 1593344

__global__ void zero_nd(float* out) {
  if (threadIdx.x == 0 && blockIdx.x == 0) out[OUT_ND] = 0.0f;
}

__global__ void fin_token(float* out, const float* E, const float* P, const float* G,
                          const float* l) {
  int i = blockIdx.x * blockDim.x + threadIdx.x;
  if (i >= S) return;
  float lv = l[i];
  float af = (lv > THRESH) ? 1.0f : 0.0f;
  out[OUT_EE + i] = E[i] * af;
  out[OUT_P + i]  = P[i] * af;
  out[OUT_G + i]  = G[i] * af;
  out[OUT_L + i]  = lv * af;
  out[OUT_A + i]  = af;
  if (af == 0.0f) atomicAdd(&out[OUT_ND], 1.0f);
}

__global__ void fin_e(float* out, const float* e, const float* l) {
  int i = blockIdx.x * blockDim.x + threadIdx.x;
  if (i >= S * DE) return;
  float af = (l[i >> 8] > THRESH) ? 1.0f : 0.0f;
  out[OUT_E + i] = e[i] * af;
}

__global__ void fin_mc(float* out, const float* m, const float* c, const float* l) {
  int i = blockIdx.x * blockDim.x + threadIdx.x;
  if (i >= S * DM) return;
  float af = (l[i >> 6] > THRESH) ? 1.0f : 0.0f;
  out[OUT_M + i] = m[i] * af;
  out[OUT_C + i] = c[i] * af;
}

// ---------------- host orchestration ----------------
extern "C" void kernel_launch(void* const* d_in, const int* in_sizes, int n_in,
                              void* d_out, int out_size, void* d_ws, size_t ws_size,
                              hipStream_t stream) {
  (void)in_sizes; (void)n_in; (void)out_size; (void)ws_size;
  const float* e_in = (const float*)d_in[0];
  const float* E_in = (const float*)d_in[1];
  const float* P_in = (const float*)d_in[2];
  const float* G_in = (const float*)d_in[3];
  const float* m_in = (const float*)d_in[4];
  const float* c_in = (const float*)d_in[5];
  const float* l_in = (const float*)d_in[6];
  const float* wq = (const float*)d_in[7];
  const float* bq = (const float*)d_in[8];
  const float* wk = (const float*)d_in[9];
  const float* bk = (const float*)d_in[10];
  const float* wv = (const float*)d_in[11];
  const float* bv = (const float*)d_in[12];
  const float* alpha = (const float*)d_in[13];
  const float* beta = (const float*)d_in[14];
  const float* gamma = (const float*)d_in[15];
  const float* basal = (const float*)d_in[16];
  float* out = (float*)d_out;

  char* ws = (char*)d_ws;
  float* e_ws   = (float*)(ws + 0);            // 4 MB
  float* h_ws   = (float*)(ws + 4194304);      // 4 MB
  float* m_ws   = (float*)(ws + 8388608);      // 1 MB
  float* E_ws   = (float*)(ws + 9437184);
  float* P_ws   = (float*)(ws + 9453568);
  float* G_ws   = (float*)(ws + 9469952);
  float* l_ws   = (float*)(ws + 9486336);
  __bf16* x_bf  = (__bf16*)(ws + 9502720);     // 2.5 MB
  __bf16* m_bf  = (__bf16*)(ws + 12124160);    // 0.5 MB
  __bf16* q_bf  = (__bf16*)(ws + 12648448);    // 2 MB
  __bf16* k_bf  = (__bf16*)(ws + 14745600);    // 2 MB
  __bf16* vt_bf = (__bf16*)(ws + 16842752);    // 2 MB (transposed V)
  __bf16* w_bf  = (__bf16*)(ws + 18939904);    // 480 KB (transposed weights)
  float* Opart  = (float*)(ws + 19431424);     // 16 MB
  float* rmaxp  = (float*)(ws + 36208640);     // 64 KB
  float* rsump  = (float*)(ws + 36274176);     // 64 KB

  const int TB = 256;
  auto blk = [](int n, int t) { return (n + t - 1) / t; };

  // working copies of mutable state
  copy_f32<<<blk(S * DE, TB), TB, 0, stream>>>(e_ws, e_in, S * DE);
  copy_f32<<<blk(S, TB), TB, 0, stream>>>(E_ws, E_in, S);
  copy_f32<<<blk(S, TB), TB, 0, stream>>>(P_ws, P_in, S);
  copy_f32<<<blk(S, TB), TB, 0, stream>>>(G_ws, G_in, S);
  copy_f32<<<blk(S * DM, TB), TB, 0, stream>>>(m_ws, m_in, S * DM);
  copy_f32<<<blk(S, TB), TB, 0, stream>>>(l_ws, l_in, S);

  for (int layer = 0; layer < DEPTH; ++layer) {
    prep_x<<<blk(S * DIN, TB), TB, 0, stream>>>(x_bf, e_ws, c_in);
    prep_bf<<<blk(S * DM, TB), TB, 0, stream>>>(m_bf, m_ws, S * DM);
    prep_w<<<blk(3 * DIN * DE, TB), TB, 0, stream>>>(w_bf, wq, wk, wv, layer);
    qkv_kernel<<<dim3(S / 16, 3), 32, 0, stream>>>(x_bf, w_bf, bq, bk, bv,
                                                   q_bf, k_bf, vt_bf, layer);
    attn_kernel<<<dim3(S / 16, KSPLIT), 32, 0, stream>>>(q_bf, k_bf, vt_bf, m_bf,
                                                         l_ws, Opart, rmaxp, rsump);
    merge_attn<<<blk(S * DE, TB), TB, 0, stream>>>(h_ws, Opart, rmaxp, rsump);
    ode_kernel<<<S, 32, 0, stream>>>(e_ws, h_ws, E_ws, P_ws, G_ws, m_ws, l_ws,
                                     alpha, beta, gamma, basal, layer);
  }

  zero_nd<<<1, 32, 0, stream>>>(out);
  fin_token<<<blk(S, TB), TB, 0, stream>>>(out, E_ws, P_ws, G_ws, l_ws);
  fin_e<<<blk(S * DE, TB), TB, 0, stream>>>(out, e_ws, l_ws);
  fin_mc<<<blk(S * DM, TB), TB, 0, stream>>>(out, m_ws, c_in, l_ws);
}
